// MixTemporalGNN_SSL_85590108274697
// MI455X (gfx1250) — matrix operand order repaired and checked by reading, not compiled, not run on hardware
//
#include <hip/hip_runtime.h>
#include <hip/hip_bf16.h>

// ---------------------------------------------------------------------------
// MixTemporalGNN forward for MI455X (gfx1250), wave32 + bf16 WMMA.
//
//  - All GEMMs: v_wmma_f32_16x16x32_bf16, bf16 activations end-to-end,
//    A-fragment loaded once per 4 N-tiles, fp32 accumulation.
//  - GCN GEMM: weight tiles staged to LDS via global_load_async_to_lds_b128
//    (ASYNCcnt) once per block; B-fragments served by ds_load; A-fragments
//    software-pipelined; BatchNorm batch stats fused into the epilogue
//    (shfl + LDS ds_add_f32 + one global atomic per column per block).
//  - LSTM weights bf16 (~84MB, L2-resident on 192MB L2), streamed per step.
//  - Edge aggregation: bf16 gather + global_atomic_add_f32 scatter.
// ---------------------------------------------------------------------------

typedef __attribute__((ext_vector_type(16))) __bf16 v16bf;
typedef __attribute__((ext_vector_type(8)))  float  v8f;

#define NN   96000
#define EE   768000
#define NGR  960

__device__ __forceinline__ __bf16 f2bf(float f) {
  union { float f; unsigned int u; } x; x.f = f;
  unsigned int r = x.u + 0x7FFFu + ((x.u >> 16) & 1u);   // round-to-nearest-even
  union { unsigned short s; __bf16 b; } y; y.s = (unsigned short)(r >> 16);
  return y.b;
}

__device__ __forceinline__ float bf2f(__bf16 b) {
  union { __bf16 b; unsigned short s; } x; x.b = b;
  union { unsigned int u; float f; } y; y.u = ((unsigned int)x.s) << 16;
  return y.f;
}

__device__ __forceinline__ float sigm(float x) { return 1.0f / (1.0f + __expf(-x)); }

__device__ __forceinline__ v8f wmma_bf16(v16bf a, v16bf b, v8f c) {
  return __builtin_amdgcn_wmma_f32_16x16x32_bf16(false, a, false, b, (short)0, c, false, false);
}

// A-fragment (16x32 bf16), ISA 16-bit A layout:
// lanes 0-15: row=lane, K in {kb..kb+7, kb+16..kb+23}; lanes 16-31: kb += 8.
__device__ __forceinline__ v16bf load_a_bf16(const __bf16* __restrict__ A, int lda,
                                             int arow, int kbA) {
  const __bf16* r = A + (size_t)arow * lda + kbA;
  v16bf a;
#pragma unroll
  for (int v = 0; v < 8; ++v) {
    int k = (v < 4) ? (2 * v) : (16 + 2 * (v - 4));
    a[2 * v]     = r[k];
    a[2 * v + 1] = r[k + 1];
  }
  return a;
}

// B-fragment (32x16 bf16) from pre-transposed weights WT[N][K]:
// lanes 0-15: col=lane, K in {k0..k0+7, k0+16..k0+23}; lanes 16-31: k0 += 8 (kbB = k0 + 16*half).
__device__ __forceinline__ v16bf load_b_bf16(const __bf16* __restrict__ WT, int K,
                                             int col, int kbB) {
  const __bf16* r = WT + (size_t)col * K + (kbB - ((kbB & 16) >> 1));
  v16bf b;
#pragma unroll
  for (int v = 0; v < 8; ++v) {
    int k = (v < 4) ? (2 * v) : (16 + 2 * (v - 4));
    b[2 * v]     = r[k];
    b[2 * v + 1] = r[k + 1];
  }
  return b;
}

// async 16B copy global -> LDS (fire-and-forget, tracked by ASYNCcnt)
__device__ __forceinline__ void async_copy_b128(unsigned lds_off, const void* gptr) {
  asm volatile("global_load_async_to_lds_b128 %0, %1, off"
               :: "v"(lds_off), "v"(gptr) : "memory");
}
__device__ __forceinline__ void wait_asynccnt0() {
  asm volatile("s_wait_asynccnt 0x0" ::: "memory");
}

// ------------------------------- conversion / graph kernels -----------------

__global__ void convT_kernel(const float* __restrict__ W, __bf16* __restrict__ WT,
                             int K, int N) {  // W[K][N] -> WT[N][K] bf16
  int idx = blockIdx.x * 256 + threadIdx.x;
  if (idx >= K * N) return;
  int k = idx / N, n = idx % N;
  WT[(size_t)n * K + k] = f2bf(W[idx]);
}

__global__ void conv_copy_kernel(const float* __restrict__ W, __bf16* __restrict__ O, int count) {
  int idx = blockIdx.x * 256 + threadIdx.x;
  if (idx < count) O[idx] = f2bf(W[idx]);
}

__global__ void deg_count_kernel(const int* __restrict__ dst, float* __restrict__ deg) {
  int e = blockIdx.x * 256 + threadIdx.x;
  if (e < EE) atomicAdd(&deg[dst[e]], 1.0f);
}

__global__ void deg_inv_kernel(float* __restrict__ deg) {
  int n = blockIdx.x * 256 + threadIdx.x;
  if (n < NN) deg[n] = 1.0f / fmaxf(deg[n], 1.0f);
}

__global__ void embed_kernel(const int* __restrict__ feat, const float* __restrict__ emb,
                             __bf16* __restrict__ hb) {
  int idx = blockIdx.x * 256 + threadIdx.x;   // NN*64
  if (idx >= NN * 64) return;
  int n = idx >> 6, c = idx & 63;
  hb[(size_t)n * 128 + c] = f2bf(emb[(size_t)feat[n] * 64 + c]);
}

// gather bf16 rows, scatter-add fp32 (4 channels/thread); rows are L2-resident.
__global__ void scatter_kernel(const __bf16* __restrict__ hb, const int* __restrict__ src,
                               const int* __restrict__ dst, float* __restrict__ hn, int K) {
  int chunks = K >> 2;
  long long idx = (long long)blockIdx.x * 256 + threadIdx.x;
  long long total = (long long)EE * chunks;
  if (idx >= total) return;
  int e = (int)(idx / chunks);
  int c = (int)(idx % chunks) * 4;
  int s = src[e], d = dst[e];
  const __bf16* hp = hb + (size_t)s * 128 + c;
  float* o = hn + (size_t)d * 128 + c;
  atomicAdd(o,     bf2f(hp[0]));
  atomicAdd(o + 1, bf2f(hp[1]));
  atomicAdd(o + 2, bf2f(hp[2]));
  atomicAdd(o + 3, bf2f(hp[3]));
}

// hnb = bf16(hn * deg_inv)   (folds the mean normalization into the A matrix)
__global__ void hn_scale_kernel(const float* __restrict__ hn, const float* __restrict__ deginv,
                                __bf16* __restrict__ hnb, int kshift) {
  long long idx = (long long)blockIdx.x * 256 + threadIdx.x;  // NN * K
  if (idx >= ((long long)NN << kshift)) return;
  int n = (int)(idx >> kshift);
  int c = (int)(idx & ((1 << kshift) - 1));
  hnb[(size_t)n * 128 + c] = f2bf(hn[(size_t)n * 128 + c] * deginv[n]);
}

// ------------------------------- GCN layer GEMM -----------------------------
// tmp = PReLU(hb @ WsT' + hnb @ WnT' + b) + fused BN batch-stat reduction.
// grid (NN/64, 2), block 128 (4 waves x 16 rows, 64 cols per block).
// B rows for this block staged to LDS asynchronously; A software-pipelined.
__global__ void __launch_bounds__(128)
gcn_gemm_kernel(const __bf16* __restrict__ hb, const __bf16* __restrict__ hnb,
                const __bf16* __restrict__ WsT, const __bf16* __restrict__ WnT,
                const float* __restrict__ bias, const float* __restrict__ alpha,
                float* __restrict__ out, float* __restrict__ stats, int K) {
  __shared__ __bf16 sB[2 * 64 * 128];     // [0]: Ws rows n0..n0+63, [8192]: Wn rows
  __shared__ float sSum[64], sSq[64];
  int tid = threadIdx.x;
  int wave = tid >> 5, lane = tid & 31;
  int n0 = blockIdx.y * 64;

  // ---- async stage of this block's 64 weight rows (both matrices) ----
  {
    int matBytes = 64 * K * 2;                       // 8KB (K=64) or 16KB (K=128)
    const char* gs = (const char*)(WsT + (size_t)n0 * K);
    const char* gn = (const char*)(WnT + (size_t)n0 * K);
    char* ls = (char*)sB;
    for (int ofs = tid * 16; ofs < matBytes; ofs += 128 * 16) {
      unsigned l0 = (unsigned)(size_t)(ls + ofs);
      async_copy_b128(l0, gs + ofs);
      async_copy_b128(l0 + (unsigned)(64 * 128 * 2), gn + ofs);
    }
    wait_asynccnt0();
  }
  if (tid < 64) { sSum[tid] = 0.0f; sSq[tid] = 0.0f; }
  __syncthreads();

  const __bf16* sWs = sB;
  const __bf16* sWn = sB + 64 * 128;

  int row0 = (blockIdx.x * 4 + wave) * 16;
  int arow = row0 + (lane & 15);
  int cl   = lane & 15;
  int kb3  = (lane >> 4) << 3;
  int kb4  = (lane >> 4) << 4;

  v8f acc0 = {0,0,0,0,0,0,0,0}, acc1 = acc0, acc2 = acc0, acc3 = acc0;
  v16bf a1 = load_a_bf16(hb,  128, arow, kb3);
  v16bf a2 = load_a_bf16(hnb, 128, arow, kb3);
  for (int k0 = 0; k0 < K; k0 += 32) {
    v16bf na1 = a1, na2 = a2;
    if (k0 + 32 < K) {                       // prefetch next A while computing
      na1 = load_a_bf16(hb,  128, arow, k0 + 32 + kb3);
      na2 = load_a_bf16(hnb, 128, arow, k0 + 32 + kb3);
    }
    int kbB = k0 + kb4;
    acc0 = wmma_bf16(a1, load_b_bf16(sWs, K,  0 + cl, kbB), acc0);
    acc1 = wmma_bf16(a1, load_b_bf16(sWs, K, 16 + cl, kbB), acc1);
    acc2 = wmma_bf16(a1, load_b_bf16(sWs, K, 32 + cl, kbB), acc2);
    acc3 = wmma_bf16(a1, load_b_bf16(sWs, K, 48 + cl, kbB), acc3);
    acc0 = wmma_bf16(a2, load_b_bf16(sWn, K,  0 + cl, kbB), acc0);
    acc1 = wmma_bf16(a2, load_b_bf16(sWn, K, 16 + cl, kbB), acc1);
    acc2 = wmma_bf16(a2, load_b_bf16(sWn, K, 32 + cl, kbB), acc2);
    acc3 = wmma_bf16(a2, load_b_bf16(sWn, K, 48 + cl, kbB), acc3);
    a1 = na1; a2 = na2;
  }

  int mb = row0 + ((lane >> 4) << 3);
  v8f accs[4] = {acc0, acc1, acc2, acc3};
#pragma unroll
  for (int j = 0; j < 4; ++j) {
    int col = n0 + j * 16 + cl;
    float bn = bias[col], al = alpha[col];
    float s = 0.0f, q = 0.0f;
#pragma unroll
    for (int r = 0; r < 8; ++r) {
      float v = accs[j][r] + bn;
      v = (v >= 0.0f) ? v : al * v;
      out[(size_t)(mb + r) * 128 + col] = v;
      s += v; q += v * v;
    }
    // combine the two half-wave lanes holding the same column
    s += __shfl_xor(s, 16, 32);
    q += __shfl_xor(q, 16, 32);
    if (lane < 16) {                          // LDS float atomics across waves
      atomicAdd(&sSum[j * 16 + cl], s);
      atomicAdd(&sSq[j * 16 + cl], q);
    }
  }
  __syncthreads();
  if (tid < 64) {
    atomicAdd(&stats[n0 + tid],       sSum[tid]);
    atomicAdd(&stats[128 + n0 + tid], sSq[tid]);
  }
}

__global__ void bn_finalize_kernel(const float* __restrict__ stats,
                                   const float* __restrict__ gamma,
                                   const float* __restrict__ beta,
                                   float* __restrict__ coef) {
  int c = threadIdx.x;
  float mu  = stats[c] * (1.0f / (float)NN);
  float var = stats[128 + c] * (1.0f / (float)NN) - mu * mu;
  float sc  = gamma[c] * rsqrtf(var + 1e-5f);
  coef[c] = sc;
  coef[128 + c] = beta[c] - mu * sc;
}

// normalize -> bf16 next-layer input + fused per-graph mean (atomics)
__global__ void bn_gvec_kernel(const float* __restrict__ tmp, const float* __restrict__ coef,
                               const int* __restrict__ gid, __bf16* __restrict__ hb,
                               float* __restrict__ gvec, int layer) {
  int n = blockIdx.x, c = threadIdx.x;
  float v = tmp[(size_t)n * 128 + c] * coef[c] + coef[128 + c];
  hb[(size_t)n * 128 + c] = f2bf(v);
  atomicAdd(&gvec[(size_t)gid[n] * 512 + layer * 128 + c], v * 0.01f);
}

// ------------------------------- generic bf16 GEMM --------------------------
#define MODE_NONE        0
#define MODE_PRELU_COL   1
#define MODE_PRELU_POINT 2
#define MODE_SIGMOID     3

// C = act(A @ WT' + b); A bf16 [M][lda]; grid (ceil(M/64), N/64), block 128.
__global__ void __launch_bounds__(128)
gemm_generic_kernel(const __bf16* __restrict__ A, int lda,
                    const __bf16* __restrict__ WT,
                    const float* __restrict__ bias,
                    const float* __restrict__ alpha,
                    float* __restrict__ Cf, __bf16* __restrict__ Cb,
                    int ldc, int M, int K, int mode) {
  int wave = threadIdx.x >> 5, lane = threadIdx.x & 31;
  int row0 = (blockIdx.x * 4 + wave) * 16;
  if (row0 >= M) return;
  int n0 = blockIdx.y * 64;
  int arow = row0 + (lane & 15);
  int cl   = lane & 15;
  v8f acc0 = {0,0,0,0,0,0,0,0}, acc1 = acc0, acc2 = acc0, acc3 = acc0;
  for (int k0 = 0; k0 < K; k0 += 32) {
    int kbA = k0 + ((lane >> 4) << 3);
    int kbB = k0 + ((lane >> 4) << 4);
    v16bf a = load_a_bf16(A, lda, arow, kbA);
    acc0 = wmma_bf16(a, load_b_bf16(WT, K, n0 +  0 + cl, kbB), acc0);
    acc1 = wmma_bf16(a, load_b_bf16(WT, K, n0 + 16 + cl, kbB), acc1);
    acc2 = wmma_bf16(a, load_b_bf16(WT, K, n0 + 32 + cl, kbB), acc2);
    acc3 = wmma_bf16(a, load_b_bf16(WT, K, n0 + 48 + cl, kbB), acc3);
  }
  int mb = row0 + ((lane >> 4) << 3);
  v8f accs[4] = {acc0, acc1, acc2, acc3};
#pragma unroll
  for (int j = 0; j < 4; ++j) {
    int col = n0 + j * 16 + cl;
    float bn = bias ? bias[col] : 0.0f;
#pragma unroll
    for (int r = 0; r < 8; ++r) {
      int m = mb + r;
      float v = accs[j][r] + bn;
      if (mode == MODE_PRELU_COL)        { float al = alpha[col];    v = (v >= 0.0f) ? v : al * v; }
      else if (mode == MODE_PRELU_POINT) { float al = alpha[m % 15]; v = (v >= 0.0f) ? v : al * v; }
      else if (mode == MODE_SIGMOID)     { v = sigm(v); }
      if (Cf) Cf[(size_t)m * ldc + col] = v;
      if (Cb) Cb[(size_t)m * ldc + col] = f2bf(v);
    }
  }
}

// z1 = sig1*px, z2 = sig2*hx -> LSTM input x0b[t][b][0:512|512:1024] (bf16)
__global__ void gate_concat_kernel(const float* __restrict__ s1, const float* __restrict__ px,
                                   const float* __restrict__ s2, const float* __restrict__ hx,
                                   __bf16* __restrict__ x) {
  int idx = blockIdx.x * 256 + threadIdx.x;   // 960*512
  if (idx >= NGR * 512) return;
  int row = idx >> 9, c = idx & 511;
  int b = row / 15, p = row % 15;
  size_t o = (size_t)p * 64 * 1024 + (size_t)b * 1024 + c;
  x[o]       = f2bf(s1[idx] * px[idx]);
  x[o + 512] = f2bf(s2[idx] * hx[idx]);
}

// ------------------------------- LSTM ---------------------------------------
// gates[64][4096] = x_t @ WihT' + h @ WhhT' + bih + bhh
// grid (1, 64), block 128: wave = row tile (rows 0..63), 4 col tiles per wave.
__global__ void __launch_bounds__(128)
lstm_gemm_kernel(const __bf16* __restrict__ x, int xK,
                 const __bf16* __restrict__ h,
                 const __bf16* __restrict__ WihT,
                 const __bf16* __restrict__ WhhT,
                 const float* __restrict__ bih,
                 const float* __restrict__ bhh,
                 float* __restrict__ gates) {
  int wave = threadIdx.x >> 5, lane = threadIdx.x & 31;
  int row0 = wave * 16;
  int n0 = blockIdx.y * 64;
  int arow = row0 + (lane & 15);
  int cl   = lane & 15;
  v8f acc0 = {0,0,0,0,0,0,0,0}, acc1 = acc0, acc2 = acc0, acc3 = acc0;
  for (int k0 = 0; k0 < xK; k0 += 32) {
    int kbA = k0 + ((lane >> 4) << 3);
    int kbB = k0 + ((lane >> 4) << 4);
    v16bf a = load_a_bf16(x, xK, arow, kbA);
    __builtin_prefetch(WihT + (size_t)(n0 + cl) * xK + k0 + 64, 0, 1);
    acc0 = wmma_bf16(a, load_b_bf16(WihT, xK, n0 +  0 + cl, kbB), acc0);
    acc1 = wmma_bf16(a, load_b_bf16(WihT, xK, n0 + 16 + cl, kbB), acc1);
    acc2 = wmma_bf16(a, load_b_bf16(WihT, xK, n0 + 32 + cl, kbB), acc2);
    acc3 = wmma_bf16(a, load_b_bf16(WihT, xK, n0 + 48 + cl, kbB), acc3);
  }
  for (int k0 = 0; k0 < 1024; k0 += 32) {
    int kbA = k0 + ((lane >> 4) << 3);
    int kbB = k0 + ((lane >> 4) << 4);
    v16bf a = load_a_bf16(h, 1024, arow, kbA);
    acc0 = wmma_bf16(a, load_b_bf16(WhhT, 1024, n0 +  0 + cl, kbB), acc0);
    acc1 = wmma_bf16(a, load_b_bf16(WhhT, 1024, n0 + 16 + cl, kbB), acc1);
    acc2 = wmma_bf16(a, load_b_bf16(WhhT, 1024, n0 + 32 + cl, kbB), acc2);
    acc3 = wmma_bf16(a, load_b_bf16(WhhT, 1024, n0 + 48 + cl, kbB), acc3);
  }
  int mb = row0 + ((lane >> 4) << 3);
  v8f accs[4] = {acc0, acc1, acc2, acc3};
#pragma unroll
  for (int j = 0; j < 4; ++j) {
    int col = n0 + j * 16 + cl;
    float bn = bih[col] + bhh[col];
#pragma unroll
    for (int r = 0; r < 8; ++r)
      gates[(size_t)(mb + r) * 4096 + col] = accs[j][r] + bn;
  }
}

// state update; h written bf16 (feeds next step's GEMM), c stays fp32
__global__ void lstm_cell_kernel(const float* __restrict__ gates, float* __restrict__ c,
                                 __bf16* __restrict__ h, __bf16* __restrict__ yout, int yld) {
  int idx = blockIdx.x * 256 + threadIdx.x;   // 64*1024
  if (idx >= 64 * 1024) return;
  int b = idx >> 10, j = idx & 1023;
  const float* g = gates + (size_t)b * 4096;
  float ig = sigm(g[j]);
  float fg = sigm(g[1024 + j]);
  float gg = tanhf(g[2048 + j]);
  float og = sigm(g[3072 + j]);
  float cc = fg * c[idx] + ig * gg;
  float hh = og * tanhf(cc);
  c[idx] = cc;
  __bf16 hb = f2bf(hh);
  h[idx] = hb;
  if (yout) yout[(size_t)b * yld + j] = hb;
}

__global__ void concat_rnnout_kernel(const __bf16* __restrict__ hb, const __bf16* __restrict__ hf,
                                     __bf16* __restrict__ out) {
  int idx = blockIdx.x * 256 + threadIdx.x;   // 64*2048
  if (idx >= 64 * 2048) return;
  int b = idx >> 11, c = idx & 2047;
  out[idx] = (c < 1024) ? hb[b * 1024 + c] : hf[b * 1024 + (c - 1024)];
}

__global__ void cls_kernel(const float* __restrict__ rep, const float* __restrict__ W,
                           const float* __restrict__ bias, float* __restrict__ out) {
  int b = blockIdx.x, j = threadIdx.x;
  if (j >= 20) return;
  float s = bias[j];
  for (int k = 0; k < 512; ++k) s += rep[b * 512 + k] * W[k * 20 + j];
  out[b * 20 + j] = s;
}

// ---------------------------------------------------------------------------

extern "C" void kernel_launch(void* const* d_in, const int* in_sizes, int n_in,
                              void* d_out, int out_size, void* d_ws, size_t ws_size,
                              hipStream_t stream) {
  (void)in_sizes; (void)n_in; (void)out_size; (void)ws_size;
#define PF(i) ((const float*)d_in[i])
#define PI(i) ((const int*)d_in[i])

  // params flattened jax-tree order (sorted dict keys) after the 8 graph arrays
  const float *clsW = PF(8),  *clsb = PF(9);
  const float *encW = PF(10), *encAl = PF(11), *encB = PF(12);
  const float *f1W1 = PF(13), *f1W2 = PF(14), *f1Al = PF(15), *f1b1 = PF(16), *f1b2 = PF(17);
  const float *f2W1 = PF(18), *f2W2 = PF(19), *f2Al = PF(20), *f2b1 = PF(21), *f2b2 = PF(22);
  const float *l0bWhh = PF(48), *l0bWih = PF(49), *l0bBhh = PF(50), *l0bBih = PF(51);
  const float *l0fWhh = PF(52), *l0fWih = PF(53), *l0fBhh = PF(54), *l0fBih = PF(55);
  const float *l1bWhh = PF(56), *l1bWih = PF(57), *l1bBhh = PF(58), *l1bBih = PF(59);
  const float *l1fWhh = PF(60), *l1fWih = PF(61), *l1fBhh = PF(62), *l1fBih = PF(63);

  // ---- workspace carve ----
  char* base = (char*)d_ws;
  size_t off = 0;
  auto allocF = [&](size_t n) { float* p = (float*)(base + off);  off = (off + n * 4 + 255) & ~(size_t)255; return p; };
  auto allocB = [&](size_t n) { __bf16* p = (__bf16*)(base + off); off = (off + n * 2 + 255) & ~(size_t)255; return p; };

  float* tmp   = allocF((size_t)NN * 128);
  float* hn    = allocF((size_t)NN * 128);
  float* deg   = allocF(NN);
  float* gvecH = allocF((size_t)NGR * 512);
  float* gvecP = allocF((size_t)NGR * 512);
  float* stats = allocF(256);
  float* coef  = allocF(256);
  float* s1    = allocF((size_t)NGR * 512);
  float* s2    = allocF((size_t)NGR * 512);
  float* gF0 = allocF(64 * 4096); float* gB0 = allocF(64 * 4096);
  float* gF1 = allocF(64 * 4096); float* gB1 = allocF(64 * 4096);
  float* csF0 = allocF(64 * 1024); float* csB0 = allocF(64 * 1024);
  float* csF1 = allocF(64 * 1024); float* csB1 = allocF(64 * 1024);
  float* rep  = allocF(64 * 512);

  __bf16* hb   = allocB((size_t)NN * 128);      // current layer input (bf16)
  __bf16* hnb  = allocB((size_t)NN * 128);      // scaled neighbor means (bf16)
  __bf16* gvbH = allocB((size_t)NGR * 512);
  __bf16* gvbP = allocB((size_t)NGR * 512);
  __bf16* zb   = allocB((size_t)NGR * 512);
  __bf16* x0b  = allocB((size_t)15 * 64 * 1024);
  __bf16* x1b  = allocB((size_t)15 * 64 * 2048);
  __bf16* hsF0 = allocB(64 * 1024); __bf16* hsB0 = allocB(64 * 1024);
  __bf16* hsF1 = allocB(64 * 1024); __bf16* hsB1 = allocB(64 * 1024);
  __bf16* rnnoutb = allocB(64 * 2048);

  __bf16* wgcnH = allocB(8 * 16384);            // 4 layers x {WnT, WsT}, 16384-elem slots
  __bf16* wgcnP = allocB(8 * 16384);
  __bf16* wf1a = allocB(512 * 512); __bf16* wf1b = allocB(512 * 512);
  __bf16* wf2a = allocB(512 * 512); __bf16* wf2b = allocB(512 * 512);
  __bf16* wenc = allocB((size_t)512 * 2048);
  __bf16* wl0f_ih = allocB((size_t)4096 * 1024); __bf16* wl0f_hh = allocB((size_t)4096 * 1024);
  __bf16* wl0b_ih = allocB((size_t)4096 * 1024); __bf16* wl0b_hh = allocB((size_t)4096 * 1024);
  __bf16* wl1f_ih = allocB((size_t)4096 * 2048); __bf16* wl1f_hh = allocB((size_t)4096 * 1024);
  __bf16* wl1b_ih = allocB((size_t)4096 * 2048); __bf16* wl1b_hh = allocB((size_t)4096 * 1024);

  // ---- per-tower GCN ----
  struct Tower {
    const int *feat, *src, *dst, *gid;
    const float* emb; int lb; float* gvec; __bf16* w;
  } tw[2] = {
    { PI(0), PI(1), PI(2), PI(3), PF(23), 24, gvecH, wgcnH },
    { PI(4), PI(5), PI(6), PI(7), PF(64), 65, gvecP, wgcnP },
  };

  for (int t = 0; t < 2; ++t) {
    Tower& T = tw[t];
    for (int l = 0; l < 4; ++l) {           // Wn -> slot 2l, Ws -> slot 2l+1
      int K = l ? 128 : 64;
      int nblk = (K * 128 + 255) / 256;
      convT_kernel<<<nblk, 256, 0, stream>>>(PF(T.lb + l * 6 + 0), T.w + (size_t)(l * 2) * 16384, K, 128);
      convT_kernel<<<nblk, 256, 0, stream>>>(PF(T.lb + l * 6 + 1), T.w + (size_t)(l * 2 + 1) * 16384, K, 128);
    }
    hipMemsetAsync(deg, 0, NN * 4, stream);
    deg_count_kernel<<<(EE + 255) / 256, 256, 0, stream>>>(T.dst, deg);
    deg_inv_kernel<<<(NN + 255) / 256, 256, 0, stream>>>(deg);
    embed_kernel<<<(NN * 64 + 255) / 256, 256, 0, stream>>>(T.feat, T.emb, hb);
    hipMemsetAsync(T.gvec, 0, (size_t)NGR * 512 * 4, stream);

    for (int l = 0; l < 4; ++l) {
      int K = l ? 128 : 64, ks = l ? 7 : 6;
      hipMemsetAsync(hn, 0, (size_t)NN * 128 * 4, stream);
      long long sthreads = (long long)EE * (K >> 2);
      scatter_kernel<<<(int)((sthreads + 255) / 256), 256, 0, stream>>>(hb, T.src, T.dst, hn, K);
      hn_scale_kernel<<<(int)(((long long)NN * K + 255) / 256), 256, 0, stream>>>(hn, deg, hnb, ks);
      hipMemsetAsync(stats, 0, 256 * 4, stream);
      gcn_gemm_kernel<<<dim3(NN / 64, 2), 128, 0, stream>>>(
          hb, hnb,
          T.w + (size_t)(l * 2 + 1) * 16384,  // WsT
          T.w + (size_t)(l * 2) * 16384,      // WnT
          PF(T.lb + l * 6 + 3),               // b
          PF(T.lb + l * 6 + 2),               // alpha
          tmp, stats, K);
      bn_finalize_kernel<<<1, 128, 0, stream>>>(stats,
          PF(T.lb + l * 6 + 5),               // gamma
          PF(T.lb + l * 6 + 4),               // beta
          coef);
      bn_gvec_kernel<<<NN, 128, 0, stream>>>(tmp, coef, T.gid, hb, T.gvec, l);
    }
  }

  // ---- cross-gated filter ----
  {
    int nblk = (512 * 512 + 255) / 256;
    convT_kernel<<<nblk, 256, 0, stream>>>(f1W1, wf1a, 512, 512);
    convT_kernel<<<nblk, 256, 0, stream>>>(f1W2, wf1b, 512, 512);
    convT_kernel<<<nblk, 256, 0, stream>>>(f2W1, wf2a, 512, 512);
    convT_kernel<<<nblk, 256, 0, stream>>>(f2W2, wf2b, 512, 512);
    int gblk = (NGR * 512 + 255) / 256;
    conv_copy_kernel<<<gblk, 256, 0, stream>>>(gvecH, gvbH, NGR * 512);
    conv_copy_kernel<<<gblk, 256, 0, stream>>>(gvecP, gvbP, NGR * 512);
  }
  dim3 gFlt(15, 8);
  gemm_generic_kernel<<<gFlt, 128, 0, stream>>>(gvbH, 512, wf1a, f1b1, f1Al, nullptr, zb, 512, NGR, 512, MODE_PRELU_POINT);
  gemm_generic_kernel<<<gFlt, 128, 0, stream>>>(zb,   512, wf1b, f1b2, nullptr, s1, nullptr, 512, NGR, 512, MODE_SIGMOID);
  gemm_generic_kernel<<<gFlt, 128, 0, stream>>>(gvbP, 512, wf2a, f2b1, f2Al, nullptr, zb, 512, NGR, 512, MODE_PRELU_POINT);
  gemm_generic_kernel<<<gFlt, 128, 0, stream>>>(zb,   512, wf2b, f2b2, nullptr, s2, nullptr, 512, NGR, 512, MODE_SIGMOID);
  gate_concat_kernel<<<(NGR * 512 + 255) / 256, 256, 0, stream>>>(s1, gvecP, s2, gvecH, x0b);

  // ---- LSTM weight conversion (already [4H, K] -> straight bf16 copy) ----
  {
    int c0 = 4096 * 1024, c1 = 4096 * 2048;
    conv_copy_kernel<<<(c0 + 255) / 256, 256, 0, stream>>>(l0fWih, wl0f_ih, c0);
    conv_copy_kernel<<<(c0 + 255) / 256, 256, 0, stream>>>(l0fWhh, wl0f_hh, c0);
    conv_copy_kernel<<<(c0 + 255) / 256, 256, 0, stream>>>(l0bWih, wl0b_ih, c0);
    conv_copy_kernel<<<(c0 + 255) / 256, 256, 0, stream>>>(l0bWhh, wl0b_hh, c0);
    conv_copy_kernel<<<(c1 + 255) / 256, 256, 0, stream>>>(l1fWih, wl1f_ih, c1);
    conv_copy_kernel<<<(c0 + 255) / 256, 256, 0, stream>>>(l1fWhh, wl1f_hh, c0);
    conv_copy_kernel<<<(c1 + 255) / 256, 256, 0, stream>>>(l1bWih, wl1b_ih, c1);
    conv_copy_kernel<<<(c0 + 255) / 256, 256, 0, stream>>>(l1bWhh, wl1b_hh, c0);
  }

  // ---- LSTM layer 0 (bidirectional, ys -> x1b) ----
  hipMemsetAsync(hsF0, 0, 64 * 1024 * 2, stream); hipMemsetAsync(csF0, 0, 64 * 1024 * 4, stream);
  hipMemsetAsync(hsB0, 0, 64 * 1024 * 2, stream); hipMemsetAsync(csB0, 0, 64 * 1024 * 4, stream);
  dim3 gLstm(1, 64);
  for (int t = 0; t < 15; ++t) {
    lstm_gemm_kernel<<<gLstm, 128, 0, stream>>>(x0b + (size_t)t * 64 * 1024, 1024, hsF0,
                                                wl0f_ih, wl0f_hh, l0fBih, l0fBhh, gF0);
    lstm_cell_kernel<<<256, 256, 0, stream>>>(gF0, csF0, hsF0, x1b + (size_t)t * 64 * 2048, 2048);
    int tb = 14 - t;
    lstm_gemm_kernel<<<gLstm, 128, 0, stream>>>(x0b + (size_t)tb * 64 * 1024, 1024, hsB0,
                                                wl0b_ih, wl0b_hh, l0bBih, l0bBhh, gB0);
    lstm_cell_kernel<<<256, 256, 0, stream>>>(gB0, csB0, hsB0, x1b + (size_t)tb * 64 * 2048 + 1024, 2048);
  }

  // ---- LSTM layer 1 (only final hidden states needed) ----
  hipMemsetAsync(hsF1, 0, 64 * 1024 * 2, stream); hipMemsetAsync(csF1, 0, 64 * 1024 * 4, stream);
  hipMemsetAsync(hsB1, 0, 64 * 1024 * 2, stream); hipMemsetAsync(csB1, 0, 64 * 1024 * 4, stream);
  for (int t = 0; t < 15; ++t) {
    lstm_gemm_kernel<<<gLstm, 128, 0, stream>>>(x1b + (size_t)t * 64 * 2048, 2048, hsF1,
                                                wl1f_ih, wl1f_hh, l1fBih, l1fBhh, gF1);
    lstm_cell_kernel<<<256, 256, 0, stream>>>(gF1, csF1, hsF1, nullptr, 0);
    int tb = 14 - t;
    lstm_gemm_kernel<<<gLstm, 128, 0, stream>>>(x1b + (size_t)tb * 64 * 2048, 2048, hsB1,
                                                wl1b_ih, wl1b_hh, l1bBih, l1bBhh, gB1);
    lstm_cell_kernel<<<256, 256, 0, stream>>>(gB1, csB1, hsB1, nullptr, 0);
  }
  concat_rnnout_kernel<<<(64 * 2048 + 255) / 256, 256, 0, stream>>>(hsB1, hsF1, rnnoutb);

  // ---- encoder + classifier ----
  convT_kernel<<<(2048 * 512 + 255) / 256, 256, 0, stream>>>(encW, wenc, 2048, 512);
  gemm_generic_kernel<<<dim3(1, 8), 128, 0, stream>>>(rnnoutb, 2048, wenc, encB, encAl,
                                                      rep, nullptr, 512, 64, 2048, MODE_PRELU_COL);
  cls_kernel<<<64, 32, 0, stream>>>(rep, clsW, clsb, (float*)d_out);

#undef PF
#undef PI
}
